// MultiHeadedAttention_8710193676743
// MI455X (gfx1250) — matrix operand order repaired and checked by reading, not compiled, not run on hardware
//
#include <hip/hip_runtime.h>
#include <hip/hip_bf16.h>

// ---------------------------------------------------------------------------
// Complex-valued multi-head attention for MI455X (gfx1250, wave32, WMMA).
// All GEMMs use v_wmma_f32_16x16x32_f16 (f16 in, f32 accumulate).
// Flash-attention streaming keeps the 2048x2048 complex score tensor out of HBM.
// LDS fragment traffic = 16B ds_load_b128; global->LDS staging is
// register-double-buffered so loads overlap the WMMA compute phase.
// ---------------------------------------------------------------------------

typedef __attribute__((ext_vector_type(16))) _Float16 v16h;
typedef __attribute__((ext_vector_type(8)))  _Float16 v8h;
typedef __attribute__((ext_vector_type(4)))  _Float16 v4h;
typedef __attribute__((ext_vector_type(8)))  float    v8f;

#define S_LEN   2048
#define DMODEL  512
#define NHEAD   8
#define DHEAD   64
#define BATCH   2
static constexpr size_t HBUF = (size_t)BATCH * NHEAD * S_LEN * DHEAD; // 2,097,152 halves

// ---- CDNA5 WMMA fragment helpers (16x16x32 f16, wave32) -------------------
__device__ __forceinline__ v16h load_a_frag(const _Float16* base, int ld, int lane) {
    const int grp = (lane >> 4) & 1;
    const int m   = lane & 15;
    const _Float16* p = base + m * ld + grp * 8;
    v8h lo = *(const v8h*)(p);
    v8h hi = *(const v8h*)(p + 16);
    return __builtin_shufflevector(lo, hi, 0, 1, 2, 3, 4, 5, 6, 7,
                                           8, 9, 10, 11, 12, 13, 14, 15);
}

// B (32x16, KxN): LDS tile stored K-contiguous per column: base[n*ld + k].
__device__ __forceinline__ v16h load_bT_frag(const _Float16* base, int ld, int lane) {
    const int grp = (lane >> 4) & 1;
    const int n   = lane & 15;
    const _Float16* p = base + n * ld + grp * 8;
    v8h lo = *(const v8h*)(p);
    v8h hi = *(const v8h*)(p + 16);
    return __builtin_shufflevector(lo, hi, 0, 1, 2, 3, 4, 5, 6, 7,
                                           8, 9, 10, 11, 12, 13, 14, 15);
}

__device__ __forceinline__ v8f wmma16(v16h a, v16h b, v8f c) {
    return __builtin_amdgcn_wmma_f32_16x16x32_f16(false, a, false, b, (short)0, c,
                                                  false, false);
}

__device__ __forceinline__ v4h cvt4(float4 f) {
    v4h hv;
    hv[0] = (_Float16)f.x; hv[1] = (_Float16)f.y;
    hv[2] = (_Float16)f.z; hv[3] = (_Float16)f.w;
    return hv;
}

// ---------------------------------------------------------------------------
// Kernel 1: fused QKV projections.  out = X @ W^T + b  (f32 in, f16 out)
// Output layout: [B, H, S, DK].  grid.x = 6*64*8 = 3072, block = 128 (4 waves)
// Register double-buffered staging: loads for step kk+32 fly during wmma of kk.
// ---------------------------------------------------------------------------
__global__ __launch_bounds__(128) void proj_kernel(
    const float* __restrict__ qr_in, const float* __restrict__ kr_in,
    const float* __restrict__ vr_in, const float* __restrict__ qp_in,
    const float* __restrict__ kp_in, const float* __restrict__ vp_in,
    const float* __restrict__ Wq, const float* __restrict__ bq,
    const float* __restrict__ Wk, const float* __restrict__ bk,
    const float* __restrict__ Wv, const float* __restrict__ bv,
    _Float16* __restrict__ ws)
{
    __shared__ __align__(16) _Float16 As [64][40]; // [m][k]
    __shared__ __align__(16) _Float16 BsT[64][40]; // [n][k] == W memory order

    const int bx    = blockIdx.x;
    const int t     = bx >> 9;
    const int rem   = bx & 511;
    const int m0    = (rem >> 3) * 64;
    const int n0    = (rem & 7) * 64;
    const int tid   = threadIdx.x;
    const int wave  = tid >> 5;
    const int lane  = tid & 31;

    const float *X, *W, *bias;
    switch (t) {
        case 0:  X = qr_in; W = Wq; bias = bq; break;
        case 1:  X = qp_in; W = Wq; bias = bq; break;
        case 2:  X = kr_in; W = Wk; bias = bk; break;
        case 3:  X = kp_in; W = Wk; bias = bk; break;
        case 4:  X = vr_in; W = Wv; bias = bv; break;
        default: X = vp_in; W = Wv; bias = bv; break;
    }
    _Float16* dst = ws + (size_t)t * HBUF;

    v8f acc[4];
#pragma unroll
    for (int ct = 0; ct < 4; ++ct) acc[ct] = (v8f){};

    float4 sa[4], sb[4];
    // prologue: issue loads for kk = 0
#pragma unroll
    for (int j = 0; j < 4; ++j) {
        int i = tid + j * 128;
        int mrow = i >> 3, k4 = (i & 7) * 4;
        sa[j] = *(const float4*)&X[(size_t)(m0 + mrow) * DMODEL + k4];
        sb[j] = *(const float4*)&W[(size_t)(n0 + mrow) * DMODEL + k4];
    }

    for (int kk = 0; kk < DMODEL; kk += 32) {
        // commit staged registers to LDS (f32 -> f16)
#pragma unroll
        for (int j = 0; j < 4; ++j) {
            int i = tid + j * 128;
            int mrow = i >> 3, k4 = (i & 7) * 4;
            *(v4h*)&As [mrow][k4] = cvt4(sa[j]);
            *(v4h*)&BsT[mrow][k4] = cvt4(sb[j]);
        }
        __syncthreads();

        // issue loads for the next k-step while wmma runs on this one
        if (kk + 32 < DMODEL) {
#pragma unroll
            for (int j = 0; j < 4; ++j) {
                int i = tid + j * 128;
                int mrow = i >> 3, k4 = (i & 7) * 4;
                sa[j] = *(const float4*)&X[(size_t)(m0 + mrow) * DMODEL + kk + 32 + k4];
                sb[j] = *(const float4*)&W[(size_t)(n0 + mrow) * DMODEL + kk + 32 + k4];
            }
        }

        v16h aF = load_a_frag(&As[wave * 16][0], 40, lane);
#pragma unroll
        for (int ct = 0; ct < 4; ++ct) {
            v16h bF = load_bT_frag(&BsT[ct * 16][0], 40, lane);
            acc[ct] = wmma16(aF, bF, acc[ct]);
        }
        __syncthreads();
    }

    const int grp = (lane >> 4) & 1;
    const int n   = lane & 15;
#pragma unroll
    for (int ct = 0; ct < 4; ++ct) {
        const int   e  = n0 + ct * 16 + n;
        const float bb = bias[e];
#pragma unroll
        for (int r = 0; r < 8; ++r) {
            int   mg = m0 + wave * 16 + r + 8 * grp;
            float v  = acc[ct][r] + bb;
            int   bi = mg >> 11, si = mg & (S_LEN - 1);
            int   hi = e >> 6,  dk = e & (DHEAD - 1);
            dst[(((size_t)bi * NHEAD + hi) * S_LEN + si) * DHEAD + dk] = (_Float16)v;
        }
    }
}

// ---------------------------------------------------------------------------
// Attention staging helpers (fixed 4-iteration loops, no exec masking)
// ---------------------------------------------------------------------------
__device__ __forceinline__ void stage64(_Float16 (*lds)[72], const _Float16* src, int tid) {
    const v8h* s = (const v8h*)src;
    v8h tmp[4];
#pragma unroll
    for (int j = 0; j < 4; ++j) tmp[j] = s[tid + j * 128];
#pragma unroll
    for (int j = 0; j < 4; ++j) {
        int e = (tid + j * 128) << 3;
        *(v8h*)&lds[e >> 6][e & 63] = tmp[j];
    }
}

__device__ __forceinline__ void issue_kv(v8h* stg,
        const _Float16* Kr, const _Float16* Kp,
        const _Float16* Vr, const _Float16* Vp, size_t off, int tid) {
    const v8h* a = (const v8h*)(Kr + off);
    const v8h* b = (const v8h*)(Kp + off);
    const v8h* c = (const v8h*)(Vr + off);
    const v8h* d = (const v8h*)(Vp + off);
#pragma unroll
    for (int j = 0; j < 4; ++j) {
        int i = tid + j * 128;
        stg[j]      = a[i];
        stg[4 + j]  = b[i];
        stg[8 + j]  = c[i];
        stg[12 + j] = d[i];
    }
}

__device__ __forceinline__ void commit_kv(const v8h* stg,
        _Float16 (*Krs)[72], _Float16 (*Kps)[72],
        _Float16 (*Vrt)[72], _Float16 (*Vpt)[72], int tid) {
#pragma unroll
    for (int j = 0; j < 4; ++j) {
        int i = tid + j * 128;
        int e = i << 3;
        *(v8h*)&Krs[e >> 6][e & 63] = stg[j];
        *(v8h*)&Kps[e >> 6][e & 63] = stg[4 + j];
        int key = i >> 3, dk0 = (i & 7) * 8;  // transpose V while committing
#pragma unroll
        for (int q = 0; q < 8; ++q) {
            Vrt[dk0 + q][key] = stg[8 + j][q];
            Vpt[dk0 + q][key] = stg[12 + j][q];
        }
    }
}

// ---------------------------------------------------------------------------
// Kernel 2: flash attention over complex scores.
// grid.x = B*H*(S/64) = 512, block = 128 (4 waves); each wave owns 16 q-rows.
// ---------------------------------------------------------------------------
__global__ __launch_bounds__(128) void attn_kernel(
    const _Float16* ws, _Float16* xr_out, _Float16* xp_out,
    const int* __restrict__ mask)
{
    __shared__ __align__(16) _Float16 Qs [64][72];
    __shared__ __align__(16) _Float16 Krs[64][72], Kps[64][72];   // [key][dk]
    __shared__ __align__(16) _Float16 Vrt[64][72], Vpt[64][72];   // [dk][key]
    __shared__ __align__(16) _Float16 Ps [4][16][72];             // per-wave P tile

    const int bx   = blockIdx.x;
    const int qt   = bx & 31;
    const int h    = (bx >> 5) & (NHEAD - 1);
    const int b    = bx >> 8;
    const int tid  = threadIdx.x;
    const int wave = tid >> 5;
    const int lane = tid & 31;
    const int grp  = (lane >> 4) & 1;
    const int ncol = lane & 15;

    const size_t off_bh = ((size_t)b * NHEAD + h) * S_LEN * DHEAD;
    const _Float16* Qr = ws + 0 * HBUF + off_bh;
    const _Float16* Qp = ws + 1 * HBUF + off_bh;
    const _Float16* Kr = ws + 2 * HBUF + off_bh;
    const _Float16* Kp = ws + 3 * HBUF + off_bh;
    const _Float16* Vr = ws + 4 * HBUF + off_bh;
    const _Float16* Vp = ws + 5 * HBUF + off_bh;

    // --- preload Q fragments (16 q-rows per wave, K=64 -> 2 chunks) ---
    v16h qrA[2], qpA[2], qpN[2];
    stage64(Qs, Qr + (size_t)qt * 64 * DHEAD, tid);
    __syncthreads();
#pragma unroll
    for (int ks = 0; ks < 2; ++ks)
        qrA[ks] = load_a_frag(&Qs[wave * 16][ks * 32], 72, lane);
    __syncthreads();
    stage64(Qs, Qp + (size_t)qt * 64 * DHEAD, tid);
    __syncthreads();
#pragma unroll
    for (int ks = 0; ks < 2; ++ks) {
        qpA[ks] = load_a_frag(&Qs[wave * 16][ks * 32], 72, lane);
        qpN[ks] = -qpA[ks];
    }
    __syncthreads();

    float mrow[8], lrow[8];
    v8f   accR[4], accP[4];
#pragma unroll
    for (int r = 0; r < 8; ++r) { mrow[r] = -1e30f; lrow[r] = 0.f; }
#pragma unroll
    for (int dt = 0; dt < 4; ++dt) { accR[dt] = (v8f){}; accP[dt] = (v8f){}; }

    v8h stg[16];
    issue_kv(stg, Kr, Kp, Vr, Vp, 0, tid);   // prologue loads for kt = 0

    for (int kt = 0; kt < S_LEN / 64; ++kt) {
        commit_kv(stg, Krs, Kps, Vrt, Vpt, tid);
        __syncthreads();
        if (kt + 1 < S_LEN / 64)
            issue_kv(stg, Kr, Kp, Vr, Vp, (size_t)(kt + 1) * 64 * DHEAD, tid);

        // ---- complex scores: Sr = QrKr^T - QpKp^T, Sp = QrKp^T + QpKr^T ----
        v8f st[4];
#pragma unroll
        for (int ct = 0; ct < 4; ++ct) {
            v8f sr = (v8f){}, sp = (v8f){};
#pragma unroll
            for (int ks = 0; ks < 2; ++ks) {
                v16h krB = load_bT_frag(&Krs[ct * 16][ks * 32], 72, lane);
                v16h kpB = load_bT_frag(&Kps[ct * 16][ks * 32], 72, lane);
                sr = wmma16(qrA[ks], krB, sr);
                sr = wmma16(qpN[ks], kpB, sr);   // minus Qp*Kp^T
                sp = wmma16(qrA[ks], kpB, sp);
                sp = wmma16(qpA[ks], krB, sp);
            }
            int key = kt * 64 + ct * 16 + ncol;
            int mv  = mask[b * S_LEN + key];
#pragma unroll
            for (int r = 0; r < 8; ++r) {
                float v = sqrtf(sr[r] * sr[r] + sp[r] * sp[r]) * 0.125f;
                st[ct][r] = (mv == 0) ? -1e9f : v;
            }
        }

        // ---- online softmax (rows r+8*grp; 16 lanes/group share rows) ----
#pragma unroll
        for (int r = 0; r < 8; ++r) {
            float v = fmaxf(fmaxf(st[0][r], st[1][r]), fmaxf(st[2][r], st[3][r]));
#pragma unroll
            for (int off = 8; off >= 1; off >>= 1) v = fmaxf(v, __shfl_xor(v, off, 32));
            float mnew  = fmaxf(mrow[r], v);
            float alpha = __expf(mrow[r] - mnew);
            mrow[r] = mnew;
            float rs = 0.f;
#pragma unroll
            for (int ct = 0; ct < 4; ++ct) {
                float p = __expf(st[ct][r] - mnew);
                Ps[wave][r + 8 * grp][ct * 16 + ncol] = (_Float16)p;
                rs += p;
            }
#pragma unroll
            for (int off = 8; off >= 1; off >>= 1) rs += __shfl_xor(rs, off, 32);
            lrow[r] = lrow[r] * alpha + rs;
#pragma unroll
            for (int dt = 0; dt < 4; ++dt) { accR[dt][r] *= alpha; accP[dt][r] *= alpha; }
        }

        // ---- P @ Vr, P @ Vp (V stored transposed -> vectorized B frags) ----
#pragma unroll
        for (int ks = 0; ks < 2; ++ks) {
            v16h pA = load_a_frag(&Ps[wave][0][ks * 32], 72, lane);
#pragma unroll
            for (int dt = 0; dt < 4; ++dt) {
                v16h vrB = load_bT_frag(&Vrt[dt * 16][ks * 32], 72, lane);
                v16h vpB = load_bT_frag(&Vpt[dt * 16][ks * 32], 72, lane);
                accR[dt] = wmma16(pA, vrB, accR[dt]);
                accP[dt] = wmma16(pA, vpB, accP[dt]);
            }
        }
        __syncthreads();
    }

    // ---- finalize: divide by l, write f16 context in [B,S,D] (heads concat) ----
#pragma unroll
    for (int r = 0; r < 8; ++r) {
        float inv = 1.f / lrow[r];
        int   q   = qt * 64 + wave * 16 + r + 8 * grp;
#pragma unroll
        for (int dt = 0; dt < 4; ++dt) {
            int    d   = h * DHEAD + dt * 16 + ncol;
            size_t idx = ((size_t)b * S_LEN + q) * DMODEL + d;
            xr_out[idx] = (_Float16)(accR[dt][r] * inv);
            xp_out[idx] = (_Float16)(accP[dt][r] * inv);
        }
    }
}

// ---------------------------------------------------------------------------
// Kernel 3: output projection.  out = X @ Wo^T + bo  (f16 X, f32 out)
// grid.x = 2 * 64 * 8 = 1024, block = 128
// ---------------------------------------------------------------------------
__global__ __launch_bounds__(128) void outproj_kernel(
    const _Float16* X0, const float* __restrict__ Wo,
    const float* __restrict__ bo, float* __restrict__ out)
{
    __shared__ __align__(16) _Float16 As [64][40];
    __shared__ __align__(16) _Float16 BsT[64][40];

    const int bx   = blockIdx.x;
    const int t    = bx >> 9;
    const int rem  = bx & 511;
    const int m0   = (rem >> 3) * 64;
    const int n0   = (rem & 7) * 64;
    const int tid  = threadIdx.x;
    const int wave = tid >> 5;
    const int lane = tid & 31;

    const _Float16* X = X0 + (size_t)t * HBUF;
    float*          o = out + (size_t)t * ((size_t)BATCH * S_LEN * DMODEL);

    v8f acc[4];
#pragma unroll
    for (int ct = 0; ct < 4; ++ct) acc[ct] = (v8f){};

    v8h    sx[2];
    float4 sb[4];
    // prologue
#pragma unroll
    for (int j = 0; j < 2; ++j) {
        int i = tid + j * 128;
        int m = i >> 2, k8 = (i & 3) * 8;
        sx[j] = *(const v8h*)&X[(size_t)(m0 + m) * DMODEL + k8];
    }
#pragma unroll
    for (int j = 0; j < 4; ++j) {
        int i = tid + j * 128;
        int n = i >> 3, k4 = (i & 7) * 4;
        sb[j] = *(const float4*)&Wo[(size_t)(n0 + n) * DMODEL + k4];
    }

    for (int kk = 0; kk < DMODEL; kk += 32) {
#pragma unroll
        for (int j = 0; j < 2; ++j) {
            int i = tid + j * 128;
            int m = i >> 2, k8 = (i & 3) * 8;
            *(v8h*)&As[m][k8] = sx[j];
        }
#pragma unroll
        for (int j = 0; j < 4; ++j) {
            int i = tid + j * 128;
            int n = i >> 3, k4 = (i & 7) * 4;
            *(v4h*)&BsT[n][k4] = cvt4(sb[j]);
        }
        __syncthreads();

        if (kk + 32 < DMODEL) {
#pragma unroll
            for (int j = 0; j < 2; ++j) {
                int i = tid + j * 128;
                int m = i >> 2, k8 = (i & 3) * 8;
                sx[j] = *(const v8h*)&X[(size_t)(m0 + m) * DMODEL + kk + 32 + k8];
            }
#pragma unroll
            for (int j = 0; j < 4; ++j) {
                int i = tid + j * 128;
                int n = i >> 3, k4 = (i & 7) * 4;
                sb[j] = *(const float4*)&Wo[(size_t)(n0 + n) * DMODEL + kk + 32 + k4];
            }
        }

        v16h aF = load_a_frag(&As[wave * 16][0], 40, lane);
#pragma unroll
        for (int ct = 0; ct < 4; ++ct) {
            v16h bF = load_bT_frag(&BsT[ct * 16][0], 40, lane);
            acc[ct] = wmma16(aF, bF, acc[ct]);
        }
        __syncthreads();
    }

    const int grp = (lane >> 4) & 1;
    const int n   = lane & 15;
#pragma unroll
    for (int ct = 0; ct < 4; ++ct) {
        const int   e  = n0 + ct * 16 + n;
        const float bb = bo[e];
#pragma unroll
        for (int r = 0; r < 8; ++r) {
            int mg = m0 + wave * 16 + r + 8 * grp;
            o[(size_t)mg * DMODEL + e] = acc[ct][r] + bb;
        }
    }
}

// ---------------------------------------------------------------------------
extern "C" void kernel_launch(void* const* d_in, const int* in_sizes, int n_in,
                              void* d_out, int out_size, void* d_ws, size_t ws_size,
                              hipStream_t stream) {
    const float* query_real  = (const float*)d_in[0];
    const float* key_real    = (const float*)d_in[1];
    const float* value_real  = (const float*)d_in[2];
    const float* query_phase = (const float*)d_in[3];
    const float* key_phase   = (const float*)d_in[4];
    const float* value_phase = (const float*)d_in[5];
    const int*   mask        = (const int*)d_in[6];
    const float* Wq = (const float*)d_in[7];
    const float* bq = (const float*)d_in[8];
    const float* Wk = (const float*)d_in[9];
    const float* bk = (const float*)d_in[10];
    const float* Wv = (const float*)d_in[11];
    const float* bv = (const float*)d_in[12];
    const float* Wo = (const float*)d_in[13];
    const float* bo = (const float*)d_in[14];

    _Float16* ws = (_Float16*)d_ws; // 8 x HBUF halves = 32 MB

    proj_kernel<<<3072, 128, 0, stream>>>(query_real, key_real, value_real,
                                          query_phase, key_phase, value_phase,
                                          Wq, bq, Wk, bk, Wv, bv, ws);
    attn_kernel<<<512, 128, 0, stream>>>(ws, ws + 6 * HBUF, ws + 7 * HBUF, mask);
    outproj_kernel<<<1024, 128, 0, stream>>>(ws + 6 * HBUF, Wo, bo, (float*)d_out);
}
